// GCN_30382598652233
// MI455X (gfx1250) — matrix operand order repaired and checked by reading, not compiled, not run on hardware
//
#include <hip/hip_runtime.h>
#include <hip/hip_bf16.h>
#include <math.h>

// GCN: out = log_softmax( GCNConv2( relu(GCNConv1(x)) ) )
// N=100000 nodes, E=3.2M edges, F_IN=128, H=16, C=2.

typedef __attribute__((ext_vector_type(16))) _Float16 v16h;
typedef __attribute__((ext_vector_type(8)))  float    v8f;

#define F_IN 128
#define HID  16
#define NCLS 2

// ---------------- utility fills ----------------
__global__ void fill_kernel(float* __restrict__ p, long long n, float v) {
    long long i = (long long)blockIdx.x * blockDim.x + threadIdx.x;
    if (i < n) p[i] = v;
}

// ---------------- degree: deg pre-seeded to 1.0 (self loop) ----------------
__global__ void deg_count_kernel(const int* __restrict__ dst, int E, float* __restrict__ deg) {
    int i = blockIdx.x * blockDim.x + threadIdx.x;
    if (i < E) atomicAdd(&deg[dst[i]], 1.0f);
}

__global__ void rsqrt_inplace_kernel(float* __restrict__ deg, int n) {
    int i = blockIdx.x * blockDim.x + threadIdx.x;
    if (i < n) {
        float d = deg[i];
        deg[i] = (d > 0.0f) ? rsqrtf(d) : 0.0f;
    }
}

// ---------------- GEMM1: H1 = X @ W1 via WMMA f16 ----------------
// One wave computes a 16x16 output tile (16 node rows x 16 hidden).
// K = 128 -> 4 x (16x16x32) WMMA steps.
//
// A (16x32 f16) layout (wave32): lane = half*16 + m, row M = m.
//   element e in v16h: K = (e<8 ? e : 16+(e-8)) + half*8   (within k-block of 32)
// B (32x16 f16) layout: col N = lane&15, half = lane>>4.
//   element e: K = (e<8 ? e : 16+(e-8)) + half*8
// C/D (v8f): col N = lane&15, element r: row M = r + half*8.
__global__ void __launch_bounds__(256)
gemm1_wmma_kernel(const float* __restrict__ x, const float* __restrict__ W1,
                  float* __restrict__ H1, int n) {
    int waveId  = (blockIdx.x * blockDim.x + threadIdx.x) >> 5;
    int lane    = threadIdx.x & 31;
    int tileRow = waveId * 16;
    if (tileRow >= n) return;                 // wave-uniform exit

    const int half = lane >> 4;
    const int m    = lane & 15;
    const int row  = tileRow + m;
    const bool rowOK = (row < n);

    v8f c = {};
#pragma unroll
    for (int kb = 0; kb < 4; ++kb) {
        // ---- A fragment: 16 f16 values for (row, k-block kb) ----
        v16h a;
        if (rowOK) {
            const float* xp = x + (size_t)row * F_IN + kb * 32 + half * 8;
#pragma unroll
            for (int e = 0; e < 8; ++e)  a[e]     = (_Float16)xp[e];       // K = base + e
#pragma unroll
            for (int e = 0; e < 8; ++e)  a[8 + e] = (_Float16)xp[16 + e];  // K = base + 16 + e
        } else {
#pragma unroll
            for (int e = 0; e < 16; ++e) a[e] = (_Float16)0.0f;
        }
        // ---- B fragment: W1 is (128,16) row-major; element e -> k = kb*32 + half*8 + (e<8 ? e : e+8) ----
        v16h b;
        const float* wp = W1 + (size_t)(kb * 32 + half * 8) * HID + m;
#pragma unroll
        for (int e = 0; e < 16; ++e) b[e] = (_Float16)wp[(e < 8 ? e : e + 8) * HID];

        c = __builtin_amdgcn_wmma_f32_16x16x32_f16(
                /*neg_a=*/false, a, /*neg_b=*/false, b,
                /*c_mod=*/(short)0, c, /*reuse_a=*/false, /*reuse_b=*/false);
    }
    // ---- store D: element r -> row tileRow + r + half*8, col m ----
#pragma unroll
    for (int r = 0; r < 8; ++r) {
        int orow = tileRow + r + half * 8;
        if (orow < n) H1[(size_t)orow * HID + m] = c[r];
    }
}

// ---------------- normalized scatter-add aggregation ----------------
// idx < E: real edge (src[idx] -> dst[idx]); idx >= E: self loop for node idx-E.
template <int F>
__global__ void scatter_agg_kernel(const int* __restrict__ src, const int* __restrict__ dst,
                                   int E, const float* __restrict__ dinv,
                                   const float* __restrict__ Hin, float* __restrict__ Aout,
                                   int n) {
    long long idx   = (long long)blockIdx.x * blockDim.x + threadIdx.x;
    long long total = (long long)E + n;
    if (idx >= total) return;
    int s, d;
    if (idx < E) { s = src[idx]; d = dst[idx]; }
    else         { s = d = (int)(idx - E); }
    float w = dinv[s] * dinv[d];
    const float* hp = Hin + (size_t)s * F;
    float*       op = Aout + (size_t)d * F;
#pragma unroll
    for (int j = 0; j < F; ++j) atomicAdd(op + j, hp[j] * w);
}

// ---------------- bias + relu (layer 1 epilogue) ----------------
__global__ void bias_relu_kernel(const float* __restrict__ A1, const float* __restrict__ b1,
                                 float* __restrict__ Hout, long long total) {
    long long i = (long long)blockIdx.x * blockDim.x + threadIdx.x;
    if (i < total) {
        float v = A1[i] + b1[(int)(i & (HID - 1))];
        Hout[i] = v > 0.0f ? v : 0.0f;
    }
}

// ---------------- layer 2 GEMM (16 -> 2), scalar ----------------
__global__ void gemm2_kernel(const float* __restrict__ H, const float* __restrict__ W2,
                             float* __restrict__ H2, int n) {
    int i = blockIdx.x * blockDim.x + threadIdx.x;
    if (i >= n) return;
    const float* h = H + (size_t)i * HID;
    float a0 = 0.0f, a1 = 0.0f;
#pragma unroll
    for (int j = 0; j < HID; ++j) {
        float hv = h[j];
        a0 += hv * W2[j * NCLS + 0];
        a1 += hv * W2[j * NCLS + 1];
    }
    H2[(size_t)i * NCLS + 0] = a0;
    H2[(size_t)i * NCLS + 1] = a1;
}

// ---------------- bias + log_softmax over 2 classes ----------------
__global__ void logsoftmax_kernel(const float* __restrict__ A2, const float* __restrict__ b2,
                                  float* __restrict__ out, int n) {
    int i = blockIdx.x * blockDim.x + threadIdx.x;
    if (i >= n) return;
    float z0 = A2[(size_t)i * NCLS + 0] + b2[0];
    float z1 = A2[(size_t)i * NCLS + 1] + b2[1];
    float mx  = fmaxf(z0, z1);
    float lse = mx + logf(expf(z0 - mx) + expf(z1 - mx));
    out[(size_t)i * NCLS + 0] = z0 - lse;
    out[(size_t)i * NCLS + 1] = z1 - lse;
}

// ---------------- host launcher ----------------
extern "C" void kernel_launch(void* const* d_in, const int* in_sizes, int n_in,
                              void* d_out, int out_size, void* d_ws, size_t ws_size,
                              hipStream_t stream) {
    const float* x  = (const float*)d_in[0];
    const int*   ei = (const int*)  d_in[1];
    const float* W1 = (const float*)d_in[2];
    const float* b1 = (const float*)d_in[3];
    const float* W2 = (const float*)d_in[4];
    const float* b2 = (const float*)d_in[5];
    float* out = (float*)d_out;

    const int N = in_sizes[0] / F_IN;   // 100000
    const int E = in_sizes[1] / 2;      // 3200000
    const int* src = ei;
    const int* dst = ei + E;

    // workspace layout (floats): dinv[N] | H1[N*16] | A1[N*16] | H2[N*2] | A2[N*2]
    float* ws   = (float*)d_ws;
    float* dinv = ws;
    float* H1   = ws + (size_t)N;
    float* A1   = ws + (size_t)17 * N;
    float* H2   = ws + (size_t)33 * N;
    float* A2   = ws + (size_t)35 * N;

    const int B = 256;
    auto blks = [](long long n, int b) { return (unsigned)((n + b - 1) / b); };

    // 1) init: dinv <- 1.0 (self loop count), A1 <- 0, A2 <- 0
    fill_kernel<<<blks((long long)N, B), B, 0, stream>>>(dinv, N, 1.0f);
    fill_kernel<<<blks((long long)N * HID, B), B, 0, stream>>>(A1, (long long)N * HID, 0.0f);
    fill_kernel<<<blks((long long)N * NCLS, B), B, 0, stream>>>(A2, (long long)N * NCLS, 0.0f);

    // 2) degree + D^{-1/2}
    deg_count_kernel<<<blks(E, B), B, 0, stream>>>(dst, E, dinv);
    rsqrt_inplace_kernel<<<blks(N, B), B, 0, stream>>>(dinv, N);

    // 3) H1 = X @ W1  (WMMA): 1 wave per 16-row tile, 8 waves per block
    {
        int tiles  = (N + 15) / 16;                 // 6250
        int blocks = (tiles + 7) / 8;               // 8 waves/block
        gemm1_wmma_kernel<<<blocks, B, 0, stream>>>(x, W1, H1, N);
    }

    // 4) A1 = norm-scatter(H1)
    scatter_agg_kernel<HID><<<blks((long long)E + N, B), B, 0, stream>>>(
        src, dst, E, dinv, H1, A1, N);

    // 5) H1 <- relu(A1 + b1)   (overwrite H1)
    bias_relu_kernel<<<blks((long long)N * HID, B), B, 0, stream>>>(A1, b1, H1, (long long)N * HID);

    // 6) H2 = H1 @ W2
    gemm2_kernel<<<blks(N, B), B, 0, stream>>>(H1, W2, H2, N);

    // 7) A2 = norm-scatter(H2)
    scatter_agg_kernel<NCLS><<<blks((long long)E + N, B), B, 0, stream>>>(
        src, dst, E, dinv, H2, A2, N);

    // 8) out = log_softmax(A2 + b2)
    logsoftmax_kernel<<<blks(N, B), B, 0, stream>>>(A2, b2, out, N);
}
